// CrossCovarianceAttn_19086834663502
// MI455X (gfx1250) — matrix operand order repaired and checked by reading, not compiled, not run on hardware
//
#include <hip/hip_runtime.h>
#include <hip/hip_bf16.h>
#include <stdint.h>

// ---------------------------------------------------------------------------
// Cross-covariance attention (XCiT) for MI455X / gfx1250.
// B=16, N=3136, C=768, H=16, Dh=48.  All heavy GEMMs on v_wmma_f32_16x16x32_bf16.
// Workspace: q/k/v [B,H,Dh,N] bf16 (3x 77MB) + attn_out [B,N,C] bf16 (77MB)
//            + invq/invk norms (2x 48KB)  => ~308.4 MB total.
// ---------------------------------------------------------------------------

typedef __attribute__((ext_vector_type(16))) __bf16 v16bf;
typedef __attribute__((ext_vector_type(8)))  __bf16 v8bf;
typedef __attribute__((ext_vector_type(4)))  __bf16 v4bf;
typedef __attribute__((ext_vector_type(8)))  float  v8f;
typedef __attribute__((ext_vector_type(4)))  float  v4f;

union BF16x16 {
    v16bf v;
    v8bf  h[2];
    __bf16 e[16];
};

// native converts: v_cvt_pk_bf16_f32 / widening shift
__device__ __forceinline__ __bf16 f2bf(float f) { return (__bf16)f; }
__device__ __forceinline__ float  bf2f(__bf16 b) { return (float)b; }

#define LSTR 40   // LDS row stride (elements): keeps 16B alignment, spreads banks

// ---------------------------------------------------------------------------
// Kernel 1: qkv = x @ qkv_w, written directly as q/k/v in [B,H,Dh,N] bf16.
// D[m=out-channel][n=token] = sum_k qkv_w[k][m] * x[token][k]
// ---------------------------------------------------------------------------
__global__ __launch_bounds__(256)
void k1_qkv(const float* __restrict__ x, const float* __restrict__ w,
            __bf16* __restrict__ qb, __bf16* __restrict__ kb, __bf16* __restrict__ vb)
{
    __shared__ __bf16 sW[128 * LSTR];   // W^T tile: [m][k]
    __shared__ __bf16 sX[128 * LSTR];   // X  tile: [n][k]

    const int t   = threadIdx.x;
    const int wid = t >> 5;
    const int ln  = t & 31;
    const int lm  = ln & 15;
    const int g   = ln >> 4;
    const int wm  = wid >> 1;          // 0..3  (32 m each)
    const int wn  = wid & 1;           // 0..1  (64 n each)
    const int tok0 = blockIdx.x * 128; // token tile base (over B*N = 50176)
    const int m0   = blockIdx.y * 128; // out-channel tile base (over 3C = 2304)

    v8f zero = {};
    v8f acc[2][4];
    for (int i = 0; i < 2; ++i)
        for (int j = 0; j < 4; ++j) acc[i][j] = zero;

    for (int kt = 0; kt < 24; ++kt) {
        const int k0 = kt * 32;
        if (kt + 1 < 24) {
            // prefetch next K tiles (global_prefetch_b8): one 64B segment/thread
            __builtin_prefetch(&x[(size_t)(tok0 + (t >> 1)) * 768 + k0 + 32 + (t & 1) * 16], 0, 1);
            __builtin_prefetch(&w[(size_t)(k0 + 32 + (t >> 3)) * 2304 + m0 + (t & 7) * 16], 0, 1);
        }
        // stage W^T (vectorized b128 reads along m, transposed b16 LDS scatter)
        for (int i = 0; i < 4; ++i) {
            int idx = t + 256 * i;          // 0..1023
            int kk  = idx >> 5;             // 0..31
            int m   = (idx & 31) * 4;       // 0..124
            v4f vv = *(const v4f*)&w[(size_t)(k0 + kk) * 2304 + m0 + m];
            sW[(m + 0) * LSTR + kk] = f2bf(vv[0]);
            sW[(m + 1) * LSTR + kk] = f2bf(vv[1]);
            sW[(m + 2) * LSTR + kk] = f2bf(vv[2]);
            sW[(m + 3) * LSTR + kk] = f2bf(vv[3]);
        }
        // stage X (vectorized b128 reads along k, contiguous b64 LDS stores)
        for (int i = 0; i < 4; ++i) {
            int idx = t + 256 * i;          // 0..1023
            int n   = idx >> 3;             // 0..127
            int kk  = (idx & 7) * 4;        // 0..28
            v4f vv = *(const v4f*)&x[(size_t)(tok0 + n) * 768 + k0 + kk];
            v4bf bv;
            bv[0] = f2bf(vv[0]); bv[1] = f2bf(vv[1]);
            bv[2] = f2bf(vv[2]); bv[3] = f2bf(vv[3]);
            *(v4bf*)&sX[n * LSTR + kk] = bv;
        }
        __syncthreads();

        BF16x16 a[2], bfr[4];
        for (int mi = 0; mi < 2; ++mi) {
            int row = wm * 32 + mi * 16 + lm;
            a[mi].h[0] = *(const v8bf*)&sW[row * LSTR + 8 * g];
            a[mi].h[1] = *(const v8bf*)&sW[row * LSTR + 16 + 8 * g];
        }
        for (int ni = 0; ni < 4; ++ni) {
            int row = wn * 64 + ni * 16 + lm;
            bfr[ni].h[0] = *(const v8bf*)&sX[row * LSTR + 8 * g];
            bfr[ni].h[1] = *(const v8bf*)&sX[row * LSTR + 16 + 8 * g];
        }
        for (int mi = 0; mi < 2; ++mi)
            for (int ni = 0; ni < 4; ++ni)
                acc[mi][ni] = __builtin_amdgcn_wmma_f32_16x16x32_bf16(
                    false, a[mi].v, false, bfr[ni].v, (short)0, acc[mi][ni],
                    false, false);
        __syncthreads();
    }

    // scatter into q/k/v [B,H,Dh,N]; each 16-wide m-tile stays in one (s,h) slice
    __bf16* bases[3] = { qb, kb, vb };
    for (int mi = 0; mi < 2; ++mi) {
        int mg = m0 + wm * 32 + mi * 16;
        int s  = mg / 768;
        int rm = mg % 768;
        int hh = rm / 48;
        int db = rm % 48;                  // 0, 16 or 32
        __bf16* outp = bases[s];
        for (int ni = 0; ni < 4; ++ni) {
            int tg = tok0 + wn * 64 + ni * 16 + lm;   // global token
            int b  = tg / 3136;
            int nn = tg - b * 3136;
            size_t rowoff = (size_t)(b * 16 + hh) * 48;
            for (int r = 0; r < 8; ++r) {
                int d = db + r + 8 * g;
                outp[(rowoff + d) * 3136 + nn] = f2bf(acc[mi][ni][r]);
            }
        }
    }
}

// ---------------------------------------------------------------------------
// Kernel 2: per-row (b,h,d) inverse L2 norm over N=3136.
// ---------------------------------------------------------------------------
__global__ __launch_bounds__(256)
void k2_norm(const __bf16* __restrict__ src, float* __restrict__ inv)
{
    __shared__ float red[256];
    const int row = blockIdx.x;
    const __bf16* p = src + (size_t)row * 3136;
    float s = 0.f;
    for (int i = threadIdx.x; i < 3136; i += 256) {
        float v = bf2f(p[i]);
        s += v * v;
    }
    red[threadIdx.x] = s;
    __syncthreads();
    for (int off = 128; off > 0; off >>= 1) {
        if (threadIdx.x < off) red[threadIdx.x] += red[threadIdx.x + off];
        __syncthreads();
    }
    if (threadIdx.x == 0) inv[row] = 1.f / fmaxf(sqrtf(red[0]), 1e-12f);
}

// ---------------------------------------------------------------------------
// Kernel 3: per-(b,h) cross-covariance attention.
//   Phase 1: S[d][e] = sum_n q[d][n]k[e][n] (3x3 WMMA tiles, K split over waves,
//            reduced via ds_add_f32), scaled by temp*invq*invk, softmax.
//   Phase 2: out^T[n][d] = sum_e v[e][n] S[d][e] via WMMA (K=64 zero-padded).
// ---------------------------------------------------------------------------
__global__ __launch_bounds__(256)
void k3_attn(const __bf16* __restrict__ qb, const __bf16* __restrict__ kb,
             const __bf16* __restrict__ vb,
             const float* __restrict__ invq, const float* __restrict__ invk,
             const float* __restrict__ temp, __bf16* __restrict__ aout)
{
    __shared__ float  Sf[48 * 48];
    __shared__ __bf16 SL[48 * 64];   // softmax(S) as bf16, K(e) padded to 64

    const int bh = blockIdx.x;
    const int b  = bh >> 4, h = bh & 15;
    const int t  = threadIdx.x, wid = t >> 5, ln = t & 31;
    const int lm = ln & 15, g = ln >> 4;

    const __bf16* qrow = qb + (size_t)bh * 48 * 3136;
    const __bf16* krow = kb + (size_t)bh * 48 * 3136;
    const __bf16* vrow = vb + (size_t)bh * 48 * 3136;

    for (int i = t; i < 2304; i += 256) Sf[i] = 0.f;
    __syncthreads();

    // ---- Phase 1: S = qn @ kn^T (normalization folded in post-GEMM) ----
    v8f zero = {};
    v8f acc[3][3];
    for (int i = 0; i < 3; ++i)
        for (int j = 0; j < 3; ++j) acc[i][j] = zero;

    for (int kt = wid; kt < 98; kt += 8) {     // 98 chunks of K=32 over N=3136
        const int n0 = kt * 32;
        BF16x16 a[3], bfr[3];
        for (int mi = 0; mi < 3; ++mi) {       // A: rows d, K contiguous in n
            const __bf16* rp = qrow + (size_t)(mi * 16 + lm) * 3136 + n0 + 8 * g;
            a[mi].h[0] = *(const v8bf*)rp;
            a[mi].h[1] = *(const v8bf*)(rp + 16);
        }
        for (int ni = 0; ni < 3; ++ni) {       // B: cols e, K contiguous in n
            const __bf16* rp = krow + (size_t)(ni * 16 + lm) * 3136 + n0 + 8 * g;
            bfr[ni].h[0] = *(const v8bf*)rp;
            bfr[ni].h[1] = *(const v8bf*)(rp + 16);
        }
        for (int mi = 0; mi < 3; ++mi)
            for (int ni = 0; ni < 3; ++ni)
                acc[mi][ni] = __builtin_amdgcn_wmma_f32_16x16x32_bf16(
                    false, a[mi].v, false, bfr[ni].v, (short)0, acc[mi][ni],
                    false, false);
    }
    for (int mi = 0; mi < 3; ++mi)
        for (int ni = 0; ni < 3; ++ni)
            for (int r = 0; r < 8; ++r)
                atomicAdd(&Sf[(mi * 16 + r + 8 * g) * 48 + ni * 16 + lm],
                          acc[mi][ni][r]);
    __syncthreads();

    // ---- softmax over e, rows handled by threads 0..47 ----
    if (t < 48) {
        const float iq = invq[bh * 48 + t];
        const float tp = temp[h];
        float mx = -1e30f;
        for (int e = 0; e < 48; ++e) {
            float v = Sf[t * 48 + e] * tp * iq * invk[bh * 48 + e];
            Sf[t * 48 + e] = v;
            mx = fmaxf(mx, v);
        }
        float sum = 0.f;
        for (int e = 0; e < 48; ++e) {
            float ex = __expf(Sf[t * 48 + e] - mx);
            Sf[t * 48 + e] = ex;
            sum += ex;
        }
        float rs = 1.f / sum;
        for (int e = 0; e < 48; ++e) SL[t * 64 + e] = f2bf(Sf[t * 48 + e] * rs);
        for (int e = 48; e < 64; ++e) SL[t * 64 + e] = f2bf(0.f);
    }
    __syncthreads();

    // ---- Phase 2: out^T[n][d] = v^T @ S^T ----
    BF16x16 sB[3][2];
    for (int di = 0; di < 3; ++di)
        for (int kc = 0; kc < 2; ++kc) {
            const __bf16* rp = &SL[(di * 16 + lm) * 64 + kc * 32 + 8 * g];
            sB[di][kc].h[0] = *(const v8bf*)rp;
            sB[di][kc].h[1] = *(const v8bf*)(rp + 16);
        }

    const __bf16 bz = f2bf(0.f);
    for (int nt = wid; nt < 196; nt += 8) {
        const int n0 = nt * 16 + lm;           // this lane's token (A-row)
        BF16x16 a0, a1;
        for (int hh = 0; hh < 16; ++hh) {
            int e = (hh < 8) ? (8 * g + hh) : (16 + 8 * g + (hh - 8));
            a0.e[hh] = vrow[(size_t)e * 3136 + n0];
            // second K-chunk: e+32 valid only for hh<8 (e<48); pad rest with 0
            a1.e[hh] = (hh < 8) ? vrow[(size_t)(32 + e) * 3136 + n0] : bz;
        }
        for (int di = 0; di < 3; ++di) {
            v8f o = {};
            o = __builtin_amdgcn_wmma_f32_16x16x32_bf16(
                    false, a0.v, false, sB[di][0].v, (short)0, o, false, false);
            o = __builtin_amdgcn_wmma_f32_16x16x32_bf16(
                    false, a1.v, false, sB[di][1].v, (short)0, o, false, false);
            for (int r = 0; r < 8; ++r) {
                int n = nt * 16 + r + 8 * g;
                aout[((size_t)(b * 3136 + n)) * 768 + h * 48 + di * 16 + lm] =
                    f2bf(o[r]);
            }
        }
    }
}

// ---------------------------------------------------------------------------
// Kernel 4: y = attn_out @ proj_w + proj_b   (fp32 output)
// ---------------------------------------------------------------------------
__global__ __launch_bounds__(256)
void k4_proj(const __bf16* __restrict__ a_in, const float* __restrict__ w,
             const float* __restrict__ bias, float* __restrict__ y)
{
    __shared__ __bf16 sA[128 * LSTR];   // [token][k]
    __shared__ __bf16 sB[128 * LSTR];   // [c][k] (W transposed)

    const int t   = threadIdx.x;
    const int wid = t >> 5;
    const int ln  = t & 31;
    const int lm  = ln & 15;
    const int g   = ln >> 4;
    const int wm  = wid >> 1;
    const int wn  = wid & 1;
    const int tok0 = blockIdx.x * 128;
    const int c0   = blockIdx.y * 128;

    v8f zero = {};
    v8f acc[2][4];
    for (int i = 0; i < 2; ++i)
        for (int j = 0; j < 4; ++j) acc[i][j] = zero;

    for (int kt = 0; kt < 24; ++kt) {
        const int k0 = kt * 32;
        if (kt + 1 < 24) {
            if (t < 128)
                __builtin_prefetch(&a_in[(size_t)(tok0 + t) * 768 + k0 + 32], 0, 1);
            __builtin_prefetch(&w[(size_t)(k0 + 32 + (t >> 3)) * 768 + c0 + (t & 7) * 16], 0, 1);
        }
        // stage A (already bf16): b128 copies, contiguous b128 LDS stores
        for (int i = 0; i < 2; ++i) {
            int idx = t + 256 * i;          // 0..511
            int m   = idx >> 2;             // 0..127
            int kk  = (idx & 3) * 8;        // 0..24
            *(v8bf*)&sA[m * LSTR + kk] =
                *(const v8bf*)&a_in[(size_t)(tok0 + m) * 768 + k0 + kk];
        }
        // stage W^T (vectorized b128 reads along c, transposed b16 LDS scatter)
        for (int i = 0; i < 4; ++i) {
            int idx = t + 256 * i;          // 0..1023
            int kk  = idx >> 5;             // 0..31
            int c   = (idx & 31) * 4;       // 0..124
            v4f vv = *(const v4f*)&w[(size_t)(k0 + kk) * 768 + c0 + c];
            sB[(c + 0) * LSTR + kk] = f2bf(vv[0]);
            sB[(c + 1) * LSTR + kk] = f2bf(vv[1]);
            sB[(c + 2) * LSTR + kk] = f2bf(vv[2]);
            sB[(c + 3) * LSTR + kk] = f2bf(vv[3]);
        }
        __syncthreads();

        BF16x16 a[2], bb[4];
        for (int mi = 0; mi < 2; ++mi) {
            int row = wm * 32 + mi * 16 + lm;
            a[mi].h[0] = *(const v8bf*)&sA[row * LSTR + 8 * g];
            a[mi].h[1] = *(const v8bf*)&sA[row * LSTR + 16 + 8 * g];
        }
        for (int ni = 0; ni < 4; ++ni) {
            int row = wn * 64 + ni * 16 + lm;
            bb[ni].h[0] = *(const v8bf*)&sB[row * LSTR + 8 * g];
            bb[ni].h[1] = *(const v8bf*)&sB[row * LSTR + 16 + 8 * g];
        }
        for (int mi = 0; mi < 2; ++mi)
            for (int ni = 0; ni < 4; ++ni)
                acc[mi][ni] = __builtin_amdgcn_wmma_f32_16x16x32_bf16(
                    false, a[mi].v, false, bb[ni].v, (short)0, acc[mi][ni],
                    false, false);
        __syncthreads();
    }

    for (int mi = 0; mi < 2; ++mi)
        for (int ni = 0; ni < 4; ++ni) {
            int c = c0 + wn * 64 + ni * 16 + lm;
            float bv = bias[c];
            for (int r = 0; r < 8; ++r) {
                int tok = tok0 + wm * 32 + mi * 16 + r + 8 * g;
                y[(size_t)tok * 768 + c] = acc[mi][ni][r] + bv;
            }
        }
}

// ---------------------------------------------------------------------------
extern "C" void kernel_launch(void* const* d_in, const int* in_sizes, int n_in,
                              void* d_out, int out_size, void* d_ws, size_t ws_size,
                              hipStream_t stream)
{
    const float* x      = (const float*)d_in[0];   // [16,3136,768]
    const float* qkv_w  = (const float*)d_in[1];   // [768,2304]
    const float* temp   = (const float*)d_in[2];   // [16,1,1]
    const float* proj_w = (const float*)d_in[3];   // [768,768]
    const float* proj_b = (const float*)d_in[4];   // [768]
    float* y = (float*)d_out;                      // [16,3136,768] f32

    const size_t QKN = (size_t)16 * 16 * 48 * 3136;   // 38,535,168 elements
    char* p = (char*)d_ws;
    __bf16* qb = (__bf16*)p;  p += QKN * 2;
    __bf16* kb = (__bf16*)p;  p += QKN * 2;
    __bf16* vb = (__bf16*)p;  p += QKN * 2;
    __bf16* ao = (__bf16*)p;  p += QKN * 2;           // attn_out [B,N,C] bf16
    float* invq = (float*)p;  p += 12288 * 4;
    float* invk = (float*)p;  p += 12288 * 4;
    (void)in_sizes; (void)n_in; (void)out_size; (void)ws_size;

    k1_qkv<<<dim3(392, 18), 256, 0, stream>>>(x, qkv_w, qb, kb, vb);
    k2_norm<<<12288, 256, 0, stream>>>(qb, invq);
    k2_norm<<<12288, 256, 0, stream>>>(kb, invk);
    k3_attn<<<256, 256, 0, stream>>>(qb, kb, vb, invq, invk, temp, ao);
    k4_proj<<<dim3(392, 6), 256, 0, stream>>>(ao, proj_w, proj_b, y);
}
